// PrunedMultiHeadAttention_75230647157611
// MI455X (gfx1250) — matrix operand order repaired and checked by reading, not compiled, not run on hardware
//
#include <hip/hip_runtime.h>

typedef __attribute__((ext_vector_type(16))) __bf16 bf16x16;
typedef __attribute__((ext_vector_type(8)))  __bf16 bf16x8;
typedef __attribute__((ext_vector_type(8)))  float  f32x8;
typedef __attribute__((ext_vector_type(4)))  int    i32x4;

#define D_MODEL 1024
#define N_HEADS 16
#define D_HEAD  64
#define B_SZ    2
#define T_SEQ   2048
#define M_TOT   (B_SZ * T_SEQ)   // 4096 rows

#if defined(__has_builtin)
#  if __has_builtin(__builtin_amdgcn_global_load_async_to_lds_b128)
#    define ASYNC_COPY 1
#  else
#    define ASYNC_COPY 0
#  endif
#else
#  define ASYNC_COPY 0
#endif

#define AS_GLOBAL __attribute__((address_space(1)))
#define AS_LDS    __attribute__((address_space(3)))

__device__ __forceinline__ f32x8 wmma_bf16(bf16x16 a, bf16x16 b, f32x8 c) {
  return __builtin_amdgcn_wmma_f32_16x16x32_bf16(false, a, false, b, (short)0, c, false, false);
}

// A fragment (16x32 bf16): lane's row data = 16 contiguous k values starting at 16*half
__device__ __forceinline__ bf16x16 load_a_frag(const __bf16* rowPtr, int half) {
  union { bf16x16 v; bf16x8 h[2]; } u;
  u.h[0] = *(const bf16x8*)(rowPtr + half * 16);
  u.h[1] = *(const bf16x8*)(rowPtr + half * 16 + 8);
  return u.v;
}

// B fragment (32x16 bf16): lane's column data = 16 contiguous k values starting at 16*half
__device__ __forceinline__ bf16x16 load_b_frag(const __bf16* colPtr, int half) {
  return *(const bf16x16*)(colPtr + half * 16);
}

__device__ __forceinline__ void wait_async_copies() {
#if ASYNC_COPY
#  if __has_builtin(__builtin_amdgcn_s_wait_asynccnt)
  __builtin_amdgcn_s_wait_asynccnt(0);
#  else
  asm volatile("s_wait_asynccnt 0" ::: "memory");
#  endif
#endif
}

// Stage one 64(n) x 32(k) bf16 B tile (4KB) into LDS: 256 threads x 16B each.
// Layout: shbuf[n*32 + k].
__device__ __forceinline__ void stage_b_tile(const __bf16* __restrict__ W, int nBase, int k0,
                                             __bf16* shbuf, int tid) {
  const int n  = tid >> 2;        // 0..63
  const int kc = (tid & 3) * 8;   // 0,8,16,24
  const __bf16* src = W + (size_t)(nBase + n) * D_MODEL + k0 + kc;
  __bf16* dst = shbuf + n * 32 + kc;
#if ASYNC_COPY
  __builtin_amdgcn_global_load_async_to_lds_b128(
      (AS_GLOBAL i32x4*)(void*)src,
      (AS_LDS i32x4*)(void*)dst, 0, 0);
#else
  *(bf16x8*)dst = *(const bf16x8*)src;
#endif
}

// Block: 256 threads (8 waves). Block tile: 256(M) x 64(N); wave tile 32x64 (8 accumulators).
// B tile (shared by all 8 waves) double-buffered in LDS via async global->LDS copies.
__device__ __forceinline__ void wave_gemm_256x64(
    const __bf16* __restrict__ X, const __bf16* __restrict__ W,
    __bf16* Bsh /* [2][64*32] */, int mBlock, int nBase, f32x8 acc[8]) {
  const int tid  = threadIdx.x;
  const int wave = tid >> 5;
  const int lane = tid & 31;
  const int half = lane >> 4;
  const int ln   = lane & 15;
  const int mTile = mBlock + wave * 32;

  const __bf16* arow0 = X + (size_t)(mTile + ln) * D_MODEL;
  const __bf16* arow1 = X + (size_t)(mTile + 16 + ln) * D_MODEL;

  stage_b_tile(W, nBase, 0, Bsh, tid);
  wait_async_copies();
  __syncthreads();

  for (int k0 = 0; k0 < D_MODEL; k0 += 32) {
    const int cur = (k0 >> 5) & 1;
    if (k0 + 32 < D_MODEL)
      stage_b_tile(W, nBase, k0 + 32, Bsh + (cur ^ 1) * (64 * 32), tid);

    bf16x16 a0 = load_a_frag(arow0 + k0, half);
    bf16x16 a1 = load_a_frag(arow1 + k0, half);
    const __bf16* bbase = Bsh + cur * (64 * 32);
#pragma unroll
    for (int j = 0; j < 4; ++j) {
      bf16x16 b = *(const bf16x16*)(bbase + (j * 16 + ln) * 32 + 16 * half);
      acc[j]     = wmma_bf16(a0, b, acc[j]);
      acc[4 + j] = wmma_bf16(a1, b, acc[4 + j]);
    }
    wait_async_copies();
    __syncthreads();
  }
}

__global__ void cvt_f32_bf16(const float* __restrict__ in, __bf16* __restrict__ out, int n) {
  int i = blockIdx.x * blockDim.x + threadIdx.x;
  int stride = gridDim.x * blockDim.x;
  for (; i < n; i += stride) out[i] = (__bf16)in[i];
}

// QKV projection: Y = X @ W^T. Grid: (M/256, N/64, 3). Block: 256.
// Q,K stored row-major bf16 (M x D_MODEL). V stored transposed per batch: Vt[b][o][t].
__global__ __launch_bounds__(256) void qkv_kernel(
    const __bf16* __restrict__ X,
    const __bf16* __restrict__ Wq, const __bf16* __restrict__ Wk, const __bf16* __restrict__ Wv,
    __bf16* __restrict__ Qb, __bf16* __restrict__ Kb, __bf16* __restrict__ Vt) {
  __shared__ __align__(32) __bf16 Bsh[2][64 * 32];
  const int wave = threadIdx.x >> 5;
  const int lane = threadIdx.x & 31;
  const int half = lane >> 4;
  const int ln   = lane & 15;
  const int mBlock = blockIdx.x * 256;
  const int nBase  = blockIdx.y * 64;
  const int which  = blockIdx.z;
  const __bf16* W = (which == 0) ? Wq : (which == 1) ? Wk : Wv;

  const f32x8 zero = {0.f,0.f,0.f,0.f,0.f,0.f,0.f,0.f};
  f32x8 acc[8];
#pragma unroll
  for (int j = 0; j < 8; ++j) acc[j] = zero;

  wave_gemm_256x64(X, W, &Bsh[0][0], mBlock, nBase, acc);

  const int mTile = mBlock + wave * 32;
#pragma unroll
  for (int i = 0; i < 2; ++i) {
#pragma unroll
    for (int j = 0; j < 4; ++j) {
#pragma unroll
      for (int r = 0; r < 8; ++r) {
        int gm  = mTile + i * 16 + r + 8 * half;
        int col = nBase + j * 16 + ln;
        __bf16 v = (__bf16)acc[i * 4 + j][r];
        if (which == 2) {
          int bb = gm >> 11, t = gm & 2047;
          Vt[((size_t)bb * D_MODEL + col) * T_SEQ + t] = v;
        } else if (which == 0) {
          Qb[(size_t)gm * D_MODEL + col] = v;
        } else {
          Kb[(size_t)gm * D_MODEL + col] = v;
        }
      }
    }
  }
}

// Flash attention: grid (T/64, H, B), block 128 (4 waves, one 16-query tile each).
__global__ __launch_bounds__(128) void attn_kernel(
    const __bf16* __restrict__ Qb, const __bf16* __restrict__ Kb, const __bf16* __restrict__ Vt,
    const int* __restrict__ head_mask, __bf16* __restrict__ Ab) {
  __shared__ __align__(32) __bf16 Plds[4][16][32];  // per-wave P tile (16 q x 32 k)
  const int wave = threadIdx.x >> 5;
  const int lane = threadIdx.x & 31;
  const int half = lane >> 4;
  const int ln   = lane & 15;
  const int qt = blockIdx.x * 4 + wave;
  const int h  = blockIdx.y;
  const int bb = blockIdx.z;
  const int q0 = qt * 16;
  const float scale = 0.125f;  // 1/sqrt(64)

  const __bf16* qrow = Qb + ((size_t)(bb * T_SEQ + q0 + ln)) * D_MODEL + h * D_HEAD;
  bf16x16 qa0 = load_a_frag(qrow,      half);
  bf16x16 qa1 = load_a_frag(qrow + 32, half);

  const f32x8 zero = {0.f,0.f,0.f,0.f,0.f,0.f,0.f,0.f};
  f32x8 O[4];
#pragma unroll
  for (int j = 0; j < 4; ++j) O[j] = zero;
  float mrow[8], lrow[8];
#pragma unroll
  for (int r = 0; r < 8; ++r) { mrow[r] = -1e30f; lrow[r] = 0.f; }

  for (int kb = 0; kb < T_SEQ; kb += 32) {
    f32x8 s0 = zero, s1 = zero;
    {
      const __bf16* k0p = Kb + ((size_t)(bb * T_SEQ + kb + ln)) * D_MODEL + h * D_HEAD;
      const __bf16* k1p = k0p + (size_t)16 * D_MODEL;
      s0 = wmma_bf16(qa0, load_b_frag(k0p,      half), s0);
      s0 = wmma_bf16(qa1, load_b_frag(k0p + 32, half), s0);
      s1 = wmma_bf16(qa0, load_b_frag(k1p,      half), s1);
      s1 = wmma_bf16(qa1, load_b_frag(k1p + 32, half), s1);
    }
#pragma unroll
    for (int r = 0; r < 8; ++r) {
      float v0 = s0[r] * scale;
      float v1 = s1[r] * scale;
      float mx = fmaxf(v0, v1);
      mx = fmaxf(mx, __shfl_xor(mx, 1, 32));
      mx = fmaxf(mx, __shfl_xor(mx, 2, 32));
      mx = fmaxf(mx, __shfl_xor(mx, 4, 32));
      mx = fmaxf(mx, __shfl_xor(mx, 8, 32));
      float mnew  = fmaxf(mrow[r], mx);
      float alpha = __expf(mrow[r] - mnew);
      mrow[r] = mnew;
      float p0 = __expf(v0 - mnew);
      float p1 = __expf(v1 - mnew);
      float ps = p0 + p1;
      ps += __shfl_xor(ps, 1, 32);
      ps += __shfl_xor(ps, 2, 32);
      ps += __shfl_xor(ps, 4, 32);
      ps += __shfl_xor(ps, 8, 32);
      lrow[r] = lrow[r] * alpha + ps;
      Plds[wave][r + 8 * half][ln]      = (__bf16)p0;
      Plds[wave][r + 8 * half][16 + ln] = (__bf16)p1;
#pragma unroll
      for (int j = 0; j < 4; ++j) O[j][r] *= alpha;
    }
    asm volatile("s_wait_dscnt 0" ::: "memory");
    bf16x16 pa;
    {
      union { bf16x16 v; bf16x8 hh[2]; } u;
      const __bf16* pr = &Plds[wave][ln][0];
      u.hh[0] = *(const bf16x8*)(pr + half * 16);
      u.hh[1] = *(const bf16x8*)(pr + half * 16 + 8);
      pa = u.v;
    }
#pragma unroll
    for (int j = 0; j < 4; ++j) {
      const __bf16* vcol =
          Vt + ((size_t)(bb * D_MODEL + h * D_HEAD + j * 16 + ln)) * T_SEQ + kb;
      O[j] = wmma_bf16(pa, load_b_frag(vcol, half), O[j]);
    }
  }

  const float mul = (head_mask[h] != 0) ? 1.f : 0.f;
#pragma unroll
  for (int r = 0; r < 8; ++r) {
    float inv = mul / lrow[r];
    int t = q0 + r + 8 * half;
#pragma unroll
    for (int j = 0; j < 4; ++j) {
      Ab[((size_t)(bb * T_SEQ + t)) * D_MODEL + h * D_HEAD + j * 16 + ln] =
          (__bf16)(O[j][r] * inv);
    }
  }
}

// Final projection: out = attn @ Wo^T, f32 output. Grid (M/256, N/64). Block 256.
__global__ __launch_bounds__(256) void out_kernel(
    const __bf16* __restrict__ A, const __bf16* __restrict__ Wo, float* __restrict__ out) {
  __shared__ __align__(32) __bf16 Bsh[2][64 * 32];
  const int wave = threadIdx.x >> 5;
  const int lane = threadIdx.x & 31;
  const int half = lane >> 4;
  const int ln   = lane & 15;
  const int mBlock = blockIdx.x * 256;
  const int nBase  = blockIdx.y * 64;

  const f32x8 zero = {0.f,0.f,0.f,0.f,0.f,0.f,0.f,0.f};
  f32x8 acc[8];
#pragma unroll
  for (int j = 0; j < 8; ++j) acc[j] = zero;

  wave_gemm_256x64(A, Wo, &Bsh[0][0], mBlock, nBase, acc);

  const int mTile = mBlock + wave * 32;
#pragma unroll
  for (int i = 0; i < 2; ++i) {
#pragma unroll
    for (int j = 0; j < 4; ++j) {
#pragma unroll
      for (int r = 0; r < 8; ++r) {
        int gm  = mTile + i * 16 + r + 8 * half;
        int col = nBase + j * 16 + ln;
        out[(size_t)gm * D_MODEL + col] = acc[i * 4 + j][r];
      }
    }
  }
}

extern "C" void kernel_launch(void* const* d_in, const int* in_sizes, int n_in,
                              void* d_out, int out_size, void* d_ws, size_t ws_size,
                              hipStream_t stream) {
  const float* x  = (const float*)d_in[0];
  const float* Wq = (const float*)d_in[1];
  const float* Wk = (const float*)d_in[2];
  const float* Wv = (const float*)d_in[3];
  const float* Wo = (const float*)d_in[4];
  const int* head_mask = (const int*)d_in[5];
  float* out = (float*)d_out;

  const size_t nX = (size_t)M_TOT * D_MODEL;       // 4M elems
  const size_t nW = (size_t)D_MODEL * D_MODEL;     // 1M elems

  char* ws = (char*)d_ws;
  size_t off = 0;
  __bf16* xb  = (__bf16*)(ws + off); off += nX * 2;
  __bf16* wqb = (__bf16*)(ws + off); off += nW * 2;
  __bf16* wkb = (__bf16*)(ws + off); off += nW * 2;
  __bf16* wvb = (__bf16*)(ws + off); off += nW * 2;
  __bf16* wob = (__bf16*)(ws + off); off += nW * 2;
  __bf16* Qb  = (__bf16*)(ws + off); off += nX * 2;
  __bf16* Kb  = (__bf16*)(ws + off); off += nX * 2;
  __bf16* Vt  = (__bf16*)(ws + off); off += nX * 2;
  __bf16* Ab  = (__bf16*)(ws + off); off += nX * 2;
  if (ws_size < off) return;  // workspace too small (deterministic no-op)

  // 1) convert inputs to bf16
  cvt_f32_bf16<<<2048, 256, 0, stream>>>(x,  xb,  (int)nX);
  cvt_f32_bf16<<<1024, 256, 0, stream>>>(Wq, wqb, (int)nW);
  cvt_f32_bf16<<<1024, 256, 0, stream>>>(Wk, wkb, (int)nW);
  cvt_f32_bf16<<<1024, 256, 0, stream>>>(Wv, wvb, (int)nW);
  cvt_f32_bf16<<<1024, 256, 0, stream>>>(Wo, wob, (int)nW);

  // 2) Q/K/V projections (z = 0/1/2 selects weight & destination)
  qkv_kernel<<<dim3(M_TOT / 256, D_MODEL / 64, 3), 256, 0, stream>>>(
      xb, wqb, wkb, wvb, Qb, Kb, Vt);

  // 3) flash attention per (q-tile, head, batch)
  attn_kernel<<<dim3(T_SEQ / 64, N_HEADS, B_SZ), 128, 0, stream>>>(
      Qb, Kb, Vt, head_mask, Ab);

  // 4) output projection (f32)
  out_kernel<<<dim3(M_TOT / 256, D_MODEL / 64, 1), 256, 0, stream>>>(Ab, wob, out);
}
